// Decoder_35742717837842
// MI455X (gfx1250) — matrix operand order repaired and checked by reading, not compiled, not run on hardware
//
#include <hip/hip_runtime.h>

typedef __attribute__((ext_vector_type(16))) _Float16 v16h;
typedef __attribute__((ext_vector_type(8)))  _Float16 v8h;
typedef __attribute__((ext_vector_type(8)))  float    v8f;

// ---------------- problem constants ----------------
constexpr int   CB      = 64;     // batch
constexpr int   CT_IN   = 1024;
constexpr int   CT_DEC  = 200;
constexpr int   CMEMR   = 400;    // MEM_DIM * R
constexpr int   CT_MEL  = 1000;
constexpr int   CMEMD   = 80;

// ---------------- packed weight layout (halfs) ----------------
// pack format: [NT][KT][32 lanes][16 halfs];  lane l: n = nt*16+(l&15),
// k = kt*32 + 8*(l>>4) + e + (e&8)   (e = 0..15), zero-padded past K.
constexpr size_t PK_W1  = 16ull*13*512;  // prenet_w1 N=256 K=400->416
constexpr size_t PK_W2  = 8ull*8*512;    // prenet_w2 N=128 K=256
constexpr size_t PK_AIH = 48ull*12*512;  // attn_wih  N=768 K=384
constexpr size_t PK_AHH = 48ull*8*512;   // attn_whh  N=768 K=256
constexpr size_t PK_QW  = 16ull*8*512;   // query_w   N=256 K=256
constexpr size_t PK_PW  = 16ull*16*512;  // proj_w    N=256 K=512
constexpr size_t PK_DG  = 48ull*8*512;   // dec*_w*h  N=768 K=256 (x4)
constexpr size_t PK_MW  = 25ull*8*512;   // mel_w     N=400 K=256
constexpr size_t PK_ANN = 16ull*8*512;   // annot_w   N=256 K=256

constexpr size_t O_W1  = 0;
constexpr size_t O_W2  = O_W1  + PK_W1;
constexpr size_t O_AIH = O_W2  + PK_W2;
constexpr size_t O_AHH = O_AIH + PK_AIH;
constexpr size_t O_QW  = O_AHH + PK_AHH;
constexpr size_t O_PW  = O_QW  + PK_QW;
constexpr size_t O_D1I = O_PW  + PK_PW;
constexpr size_t O_D1H = O_D1I + PK_DG;
constexpr size_t O_D2I = O_D1H + PK_DG;
constexpr size_t O_D2H = O_D2I + PK_DG;
constexpr size_t O_MW  = O_D2H + PK_DG;
constexpr size_t O_ANN = O_MW  + PK_MW;
constexpr size_t PK_TOT = O_ANN + PK_ANN;           // 1,781,760 halfs

// ---------------- workspace layout (bytes) ----------------
constexpr size_t PA_HALFS = (size_t)CB * CT_IN * 256;      // proc_annots (f16)
constexpr size_t OFF_PK_B = PA_HALFS * 2;                  // 32 MB
constexpr size_t OFF_ST_B = OFF_PK_B + PK_TOT * 2;         // per-WG fp32 state

// per-workgroup state region (floats)
constexpr int ST_HA   = 0;            // attn hidden   16x256
constexpr int ST_H1   = 4096;
constexpr int ST_H2   = 8192;
constexpr int ST_CTX  = 12288;
constexpr int ST_D    = 16384;
constexpr int ST_GI   = 20480;        // 16x768
constexpr int ST_GH   = 32768;        // 16x768
constexpr int ST_TMP  = 45056;        // 16x416 scratch (pr1/pr2/q/d/mel)
constexpr int ST_STRIDE = 51712;

// ---------------- device helpers ----------------

// Write one fp32 activation element (row m, col k) into LDS directly in the
// WMMA f16 A-operand per-lane layout:
//   lane = 16*h + m  with h = (k>>3)&1 ;  element e = 8*((k>>4)&1) + (k&7)
//   stored at af16[(k>>5)*512 + lane*16 + e]
__device__ __forceinline__ void stage_a(_Float16* __restrict__ af16,
                                        int m, int k, float v) {
  const int h = (k >> 3) & 1;
  const int e = ((k >> 4) & 1) * 8 + (k & 7);
  af16[(size_t)(k >> 5) * 512 + (size_t)((h << 4) | m) * 16 + e] = (_Float16)v;
}

// 16-wave workgroup GEMM: out[16][N] = A(f16, pre-staged LDS) * packed-W.
// A fragment = one aligned 32B LDS load per kt; B fragment = one 32B global load.
__device__ __forceinline__ void wave_gemm(const _Float16* __restrict__ af16,
                                          const _Float16* __restrict__ wp,
                                          int NT, int KT,
                                          float* __restrict__ outp, int N,
                                          int wave, int lane) {
  const int m  = lane & 15;
  const int hh = lane >> 4;
  const _Float16* apt = af16 + (size_t)lane * 16;
  for (int nt = wave; nt < NT; nt += 16) {
    v8f acc = {};
    const _Float16* wpt = wp + (size_t)nt * KT * 512 + (size_t)lane * 16;
    for (int kt = 0; kt < KT; ++kt) {
      v16h a = *(const v16h*)(apt + (size_t)kt * 512);
      v16h b = *(const v16h*)(wpt + (size_t)kt * 512);
      acc = __builtin_amdgcn_wmma_f32_16x16x32_f16(false, a, false, b,
                                                   (short)0, acc, false, false);
    }
    const int n  = nt * 16 + m;
    const int mb = hh * 8;
#pragma unroll
    for (int r = 0; r < 8; ++r) outp[(size_t)(mb + r) * N + n] = acc[r];
  }
}

// torch GRUCell elementwise update (gates precomputed without biases);
// optionally restages the new hidden state as the next GEMM's A operand.
__device__ __forceinline__ void gru_update(int tid,
                                           const float* __restrict__ gi,
                                           const float* __restrict__ gh,
                                           const float* __restrict__ bih,
                                           const float* __restrict__ bhh,
                                           float* __restrict__ h,
                                           _Float16* __restrict__ af16) {
  for (int idx = tid; idx < 16 * 256; idx += 512) {
    const int m = idx >> 8, k = idx & 255;
    const float ir  = gi[m * 768 + k]       + bih[k];
    const float iz  = gi[m * 768 + 256 + k] + bih[256 + k];
    const float inn = gi[m * 768 + 512 + k] + bih[512 + k];
    const float hr  = gh[m * 768 + k]       + bhh[k];
    const float hz  = gh[m * 768 + 256 + k] + bhh[256 + k];
    const float hn  = gh[m * 768 + 512 + k] + bhh[512 + k];
    const float r = 1.f / (1.f + __expf(-(ir + hr)));
    const float z = 1.f / (1.f + __expf(-(iz + hz)));
    const float n = tanhf(inn + r * hn);
    const float hv = h[idx];
    const float hnew = (1.f - z) * n + z * hv;
    h[idx] = hnew;
    if (af16) stage_a(af16, m, k, hnew);
  }
}

// ---------------- kernel 0: weight packer ----------------
__global__ void pack_w_kernel(const float* __restrict__ W, _Float16* __restrict__ dst,
                              int N, int K, int KT, int total) {
  int idx = blockIdx.x * blockDim.x + threadIdx.x;
  if (idx >= total) return;
  const int e    = idx & 15;
  const int lane = (idx >> 4) & 31;
  const int tile = idx >> 9;
  const int kt   = tile % KT;
  const int nt   = tile / KT;
  const int n = nt * 16 + (lane & 15);
  const int k = kt * 32 + 8 * (lane >> 4) + e + (e & 8);
  dst[idx] = (n < N && k < K) ? (_Float16)W[(size_t)n * K + k] : (_Float16)0.f;
}

// ---------------- kernel 1: proc_annots = inputs @ annot_w.T + annot_b (f16 out) ----------------
__global__ __launch_bounds__(256) void annot_kernel(const float* __restrict__ inputs,
                                                    const _Float16* __restrict__ wann,
                                                    const float* __restrict__ bann,
                                                    _Float16* __restrict__ pa) {
  const int lane = threadIdx.x & 31;
  const int wave = threadIdx.x >> 5;
  const int rt   = blockIdx.x * 8 + wave;      // 4096 row tiles of 16 rows
  const int m  = lane & 15;
  const int hh = lane >> 4;
  const float* A0 = inputs + (size_t)rt * 16 * 256;
  for (int ntc = 0; ntc < 4; ++ntc) {
    v8f acc0 = {}, acc1 = {}, acc2 = {}, acc3 = {};
    for (int kt = 0; kt < 8; ++kt) {
      const float* prow = A0 + (size_t)m * 256 + kt * 32;
      v16h a;
#pragma unroll
      for (int i = 0; i < 8; ++i) {
        const int kb = (i < 4) ? (2 * i + 8 * hh) : (16 + 2 * (i - 4) + 8 * hh);
        a[2 * i]     = (_Float16)prow[kb];
        a[2 * i + 1] = (_Float16)prow[kb + 1];
      }
      const size_t lb = (size_t)lane * 16;
      v16h b0 = *(const v16h*)(wann + ((size_t)((ntc * 4 + 0) * 8 + kt) * 512) + lb);
      v16h b1 = *(const v16h*)(wann + ((size_t)((ntc * 4 + 1) * 8 + kt) * 512) + lb);
      v16h b2 = *(const v16h*)(wann + ((size_t)((ntc * 4 + 2) * 8 + kt) * 512) + lb);
      v16h b3 = *(const v16h*)(wann + ((size_t)((ntc * 4 + 3) * 8 + kt) * 512) + lb);
      acc0 = __builtin_amdgcn_wmma_f32_16x16x32_f16(false, a, false, b0, (short)0, acc0, false, false);
      acc1 = __builtin_amdgcn_wmma_f32_16x16x32_f16(false, a, false, b1, (short)0, acc1, false, false);
      acc2 = __builtin_amdgcn_wmma_f32_16x16x32_f16(false, a, false, b2, (short)0, acc2, false, false);
      acc3 = __builtin_amdgcn_wmma_f32_16x16x32_f16(false, a, false, b3, (short)0, acc3, false, false);
    }
    const int mb = hh * 8;
    const v8f* accs[4] = {&acc0, &acc1, &acc2, &acc3};
#pragma unroll
    for (int u = 0; u < 4; ++u) {
      const int n = (ntc * 4 + u) * 16 + m;
      const float bn = bann[n];
#pragma unroll
      for (int r = 0; r < 8; ++r) {
        const int row = rt * 16 + mb + r;
        pa[(size_t)row * 256 + n] = (_Float16)((*accs[u])[r] + bn);
      }
    }
  }
}

// ---------------- kernel 2: persistent 200-step decoder (1 WG per 16 batch rows) ----------------
__global__ __launch_bounds__(512, 1) void decoder_kernel(
    const float* __restrict__ inputs, const float* __restrict__ memory,
    const float* __restrict__ b1, const float* __restrict__ b2,
    const float* __restrict__ abih, const float* __restrict__ abhh,
    const float* __restrict__ qb, const float* __restrict__ vw,
    const float* __restrict__ pb,
    const float* __restrict__ d1bi, const float* __restrict__ d1bh,
    const float* __restrict__ d2bi, const float* __restrict__ d2bh,
    const float* __restrict__ melb,
    char* __restrict__ ws, float* __restrict__ out) {
  __shared__ __align__(32) _Float16 af16[16 * 512];  // staged f16 A operand (16 KB)
  __shared__ float abuf[16 * 1024];                  // alignments for context (64 KB)
  __shared__ float qbuf[16 * 256];                   // query + bias (16 KB)
  __shared__ float vbuf[256];

  const int tid  = threadIdx.x;
  const int wave = tid >> 5;
  const int lane = tid & 31;
  const int bt   = blockIdx.x;                 // batch tile 0..3

  const _Float16* pa = (const _Float16*)ws;
  const _Float16* pk = (const _Float16*)(ws + OFF_PK_B);
  float* st = (float*)(ws + OFF_ST_B) + (size_t)bt * ST_STRIDE;

  float* outs   = out;                                   // [B, 200, 400]
  float* aligns = out + (size_t)CB * CT_DEC * CMEMR;     // [B, 200, 1024]

  for (int i = tid; i < 5 * 4096; i += 512) st[i] = 0.f; // zero states
  for (int i = tid; i < 256; i += 512) vbuf[i] = vw[i];
  __syncthreads();

  for (int t = 0; t < CT_DEC; ++t) {
    // prefetch next step's teacher-forcing frames toward the WGP
    if (t + 1 < CT_DEC) {
      for (int p = tid; p < 16 * 13; p += 512) {
        const int m = p / 13, c = p % 13;
        const char* adr = (const char*)(memory +
            (size_t)(bt * 16 + m) * CT_MEL * CMEMD + (size_t)t * CMEMR) + c * 128;
        __builtin_prefetch(adr, 0, 1);
      }
    }
    // -- prenet layer 1: m_in [16,416(pad)] -> 256
    for (int idx = tid; idx < 16 * 416; idx += 512) {
      const int m = idx / 416, k = idx - m * 416;
      float v = 0.f;
      if (t > 0 && k < CMEMR)
        v = memory[(size_t)(bt * 16 + m) * CT_MEL * CMEMD + (size_t)(t - 1) * CMEMR + k];
      stage_a(af16, m, k, v);
    }
    __syncthreads();
    wave_gemm(af16, pk + O_W1, 16, 13, st + ST_TMP, 256, wave, lane);
    __syncthreads();
    for (int idx = tid; idx < 4096; idx += 512) {
      const int m = idx >> 8, k = idx & 255;
      const float v = st[ST_TMP + idx] + b1[k];
      stage_a(af16, m, k, v > 0.f ? v : 0.f);
    }
    __syncthreads();
    // -- prenet layer 2: 256 -> 128
    wave_gemm(af16, pk + O_W2, 8, 8, st + ST_TMP, 128, wave, lane);
    __syncthreads();
    // -- attention GRU input x = cat(relu(pr2), ctx)  [16,384]
    for (int idx = tid; idx < 16 * 384; idx += 512) {
      const int m = idx / 384, k = idx - m * 384;
      float v;
      if (k < 128) { const float p = st[ST_TMP + m * 128 + k] + b2[k]; v = p > 0.f ? p : 0.f; }
      else v = st[ST_CTX + m * 256 + (k - 128)];
      stage_a(af16, m, k, v);
    }
    __syncthreads();
    wave_gemm(af16, pk + O_AIH, 48, 12, st + ST_GI, 768, wave, lane);
    __syncthreads();
    for (int idx = tid; idx < 4096; idx += 512)
      stage_a(af16, idx >> 8, idx & 255, st[ST_HA + idx]);
    __syncthreads();
    wave_gemm(af16, pk + O_AHH, 48, 8, st + ST_GH, 768, wave, lane);
    __syncthreads();
    gru_update(tid, st + ST_GI, st + ST_GH, abih, abhh, st + ST_HA, af16);
    __syncthreads();
    // -- query projection
    wave_gemm(af16, pk + O_QW, 16, 8, st + ST_TMP, 256, wave, lane);
    __syncthreads();
    for (int idx = tid; idx < 4096; idx += 512)
      qbuf[idx] = st[ST_TMP + idx] + qb[idx & 255];
    __syncthreads();
    // -- Bahdanau attention: one wave per batch row, 32 energies per lane
    {
      const int row = wave;
      const int bg  = bt * 16 + row;
      const float* qr = qbuf + row * 256;
      float ev[32];
#pragma unroll
      for (int j = 0; j < 32; ++j) {
        const int ti = lane + j * 32;
        const _Float16* pr = pa + ((size_t)bg * CT_IN + ti) * 256;
        float dot = 0.f;
        for (int k = 0; k < 256; k += 8) {
          v8h c = *(const v8h*)(pr + k);
#pragma unroll
          for (int u = 0; u < 8; ++u)
            dot += vbuf[k + u] * tanhf((float)c[u] + qr[k + u]);
        }
        ev[j] = dot;
      }
      float mx = ev[0];
#pragma unroll
      for (int j = 1; j < 32; ++j) mx = fmaxf(mx, ev[j]);
      for (int off = 16; off > 0; off >>= 1) mx = fmaxf(mx, __shfl_xor(mx, off, 32));
      float sum = 0.f;
#pragma unroll
      for (int j = 0; j < 32; ++j) { ev[j] = __expf(ev[j] - mx); sum += ev[j]; }
      for (int off = 16; off > 0; off >>= 1) sum += __shfl_xor(sum, off, 32);
      const float inv = 1.f / sum;
      float* arow = abuf + row * 1024;
      float* orow = aligns + ((size_t)bg * CT_DEC + t) * 1024;
#pragma unroll
      for (int j = 0; j < 32; ++j) {
        const int ti = lane + j * 32;
        const float al = ev[j] * inv;
        arow[ti] = al;
        orow[ti] = al;
      }
    }
    __syncthreads();
    // -- context = align @ inputs
    for (int idx = tid; idx < 4096; idx += 512) {
      const int m = idx >> 8, k = idx & 255;
      const float* ip = inputs + (size_t)(bt * 16 + m) * CT_IN * 256 + k;
      const float* ar = abuf + m * 1024;
      float acc = 0.f;
      for (int ti = 0; ti < CT_IN; ++ti) acc += ar[ti] * ip[(size_t)ti * 256];
      st[ST_CTX + idx] = acc;
    }
    __syncthreads();
    // -- decoder input projection: cat(attn_h, ctx) [16,512] -> 256
    for (int idx = tid; idx < 16 * 512; idx += 512) {
      const int m = idx >> 9, k = idx & 511;
      const float v = (k < 256) ? st[ST_HA + m * 256 + k]
                                : st[ST_CTX + m * 256 + (k - 256)];
      stage_a(af16, m, k, v);
    }
    __syncthreads();
    wave_gemm(af16, pk + O_PW, 16, 16, st + ST_TMP, 256, wave, lane);
    __syncthreads();
    for (int idx = tid; idx < 4096; idx += 512) {
      const float dv = st[ST_TMP + idx] + pb[idx & 255];
      st[ST_D + idx] = dv;
      stage_a(af16, idx >> 8, idx & 255, dv);
    }
    __syncthreads();
    // -- residual GRU 1
    wave_gemm(af16, pk + O_D1I, 48, 8, st + ST_GI, 768, wave, lane);
    __syncthreads();
    for (int idx = tid; idx < 4096; idx += 512)
      stage_a(af16, idx >> 8, idx & 255, st[ST_H1 + idx]);
    __syncthreads();
    wave_gemm(af16, pk + O_D1H, 48, 8, st + ST_GH, 768, wave, lane);
    __syncthreads();
    gru_update(tid, st + ST_GI, st + ST_GH, d1bi, d1bh, st + ST_H1, nullptr);
    __syncthreads();
    for (int idx = tid; idx < 4096; idx += 512) {
      const float dv = st[ST_H1 + idx] + st[ST_D + idx];
      st[ST_D + idx] = dv;
      stage_a(af16, idx >> 8, idx & 255, dv);
    }
    __syncthreads();
    // -- residual GRU 2
    wave_gemm(af16, pk + O_D2I, 48, 8, st + ST_GI, 768, wave, lane);
    __syncthreads();
    for (int idx = tid; idx < 4096; idx += 512)
      stage_a(af16, idx >> 8, idx & 255, st[ST_H2 + idx]);
    __syncthreads();
    wave_gemm(af16, pk + O_D2H, 48, 8, st + ST_GH, 768, wave, lane);
    __syncthreads();
    gru_update(tid, st + ST_GI, st + ST_GH, d2bi, d2bh, st + ST_H2, nullptr);
    __syncthreads();
    for (int idx = tid; idx < 4096; idx += 512) {
      const float dv = st[ST_H2 + idx] + st[ST_D + idx];
      st[ST_D + idx] = dv;
      stage_a(af16, idx >> 8, idx & 255, dv);
    }
    __syncthreads();
    // -- mel projection 256 -> 400
    wave_gemm(af16, pk + O_MW, 25, 8, st + ST_TMP, 400, wave, lane);
    __syncthreads();
    for (int idx = tid; idx < 16 * 400; idx += 512) {
      const int m = idx / 400, k = idx - m * 400;
      const int bg = bt * 16 + m;
      outs[((size_t)bg * CT_DEC + t) * 400 + k] = st[ST_TMP + m * 400 + k] + melb[k];
    }
    __syncthreads();
  }
}

// ---------------- host launcher ----------------
extern "C" void kernel_launch(void* const* d_in, const int* in_sizes, int n_in,
                              void* d_out, int out_size, void* d_ws, size_t ws_size,
                              hipStream_t stream) {
  (void)in_sizes; (void)n_in; (void)out_size; (void)ws_size;
  const float* inputs = (const float*)d_in[0];
  const float* memory = (const float*)d_in[1];
  const float* w1   = (const float*)d_in[2];
  const float* b1   = (const float*)d_in[3];
  const float* w2   = (const float*)d_in[4];
  const float* b2   = (const float*)d_in[5];
  const float* aih  = (const float*)d_in[6];
  const float* ahh  = (const float*)d_in[7];
  const float* abih = (const float*)d_in[8];
  const float* abhh = (const float*)d_in[9];
  const float* qw   = (const float*)d_in[10];
  const float* qb   = (const float*)d_in[11];
  const float* annw = (const float*)d_in[12];
  const float* annb = (const float*)d_in[13];
  const float* vw   = (const float*)d_in[14];
  const float* pw   = (const float*)d_in[15];
  const float* pb   = (const float*)d_in[16];
  const float* d1i  = (const float*)d_in[17];
  const float* d1h  = (const float*)d_in[18];
  const float* d1bi = (const float*)d_in[19];
  const float* d1bh = (const float*)d_in[20];
  const float* d2i  = (const float*)d_in[21];
  const float* d2h  = (const float*)d_in[22];
  const float* d2bi = (const float*)d_in[23];
  const float* d2bh = (const float*)d_in[24];
  const float* melw = (const float*)d_in[25];
  const float* melb = (const float*)d_in[26];

  char* ws = (char*)d_ws;
  _Float16* pa = (_Float16*)ws;
  _Float16* pk = (_Float16*)(ws + OFF_PK_B);

  auto pack = [&](const float* W, size_t off, int N, int K, int NT, int KT) {
    const int total = NT * KT * 512;
    pack_w_kernel<<<(total + 255) / 256, 256, 0, stream>>>(W, pk + off, N, K, KT, total);
  };
  pack(w1,   O_W1,  256, 400, 16, 13);
  pack(w2,   O_W2,  128, 256,  8,  8);
  pack(aih,  O_AIH, 768, 384, 48, 12);
  pack(ahh,  O_AHH, 768, 256, 48,  8);
  pack(qw,   O_QW,  256, 256, 16,  8);
  pack(pw,   O_PW,  256, 512, 16, 16);
  pack(d1i,  O_D1I, 768, 256, 48,  8);
  pack(d1h,  O_D1H, 768, 256, 48,  8);
  pack(d2i,  O_D2I, 768, 256, 48,  8);
  pack(d2h,  O_D2H, 768, 256, 48,  8);
  pack(melw, O_MW,  400, 256, 25,  8);
  pack(annw, O_ANN, 256, 256, 16,  8);

  // hoisted annotation projection -> f16 proc_annots (WMMA)
  annot_kernel<<<512, 256, 0, stream>>>(inputs, pk + O_ANN, annb, pa);

  // persistent 200-step decoder, one workgroup per 16 batch rows
  decoder_kernel<<<4, 512, 0, stream>>>(inputs, memory, b1, b2, abih, abhh,
                                        qb, vw, pb, d1bi, d1bh, d2bi, d2bh,
                                        melb, ws, (float*)d_out);
}
